// SMC_Transf_Cell_29205777612850
// MI455X (gfx1250) — matrix operand order repaired and checked by reading, not compiled, not run on hardware
//
#include <hip/hip_runtime.h>
#include <hip/hip_bf16.h>

typedef __attribute__((ext_vector_type(16))) _Float16 v16h;
typedef __attribute__((ext_vector_type(8)))  _Float16 v8h;
typedef __attribute__((ext_vector_type(8)))  float    v8f;

#define BB   16
#define PP   25
#define SS   1024
#define DD   128
#define HH   8
#define DHH  16
#define DFF  512
#define SIGMA_ 0.1f
#define OMEGA_ 0.5f
#define NPART (BB*PP)   // 400 rows = exactly 25 WMMA M-tiles

// ---------------------------------------------------------------------------
// WMMA fragment loaders (layouts per CDNA5 ISA 7.12.2, wave32)
// ---------------------------------------------------------------------------
__device__ __forceinline__ v16h load_a_tile(const _Float16* __restrict__ A, int lda,
                                            int m0, int k0, int lane) {
  // 16-bit A 16x32: lanes 0-15 hold row M=lane, K=0..7 / 16..23
  //                 lanes 16-31 hold row M=lane-16, K=8..15 / 24..31
  const int r  = lane & 15;
  const int kh = (lane >> 4) * 8;
  const _Float16* p = A + (size_t)(m0 + r) * lda + (k0 + kh);
  const v8h a0 = *(const v8h*)p;          // 16B aligned: kh,k0 multiples of 8
  const v8h a1 = *(const v8h*)(p + 16);
  return __builtin_shufflevector(a0, a1, 0,1,2,3,4,5,6,7,8,9,10,11,12,13,14,15);
}

// B fragment from PRE-TRANSPOSED weights: BT is N x K row-major (ldk = K).
// 16-bit B 32x16: lane = column; lanes 0-15 K=0..7 / 16..23, lanes 16-31 K=8..15 / 24..31.
__device__ __forceinline__ v16h load_b_tileT(const _Float16* __restrict__ BT, int ldk,
                                             int k0, int n0, int lane) {
  const int c  = lane & 15;
  const int kb = (lane >> 4) * 8;
  const _Float16* p = BT + (size_t)(n0 + c) * ldk + (k0 + kb);
  const v8h b0 = *(const v8h*)p;          // K runs kb..kb+7 and kb+16..kb+23 -> two B128 loads
  const v8h b1 = *(const v8h*)(p + 16);
  return __builtin_shufflevector(b0, b1, 0,1,2,3,4,5,6,7,8,9,10,11,12,13,14,15);
}

// One M-tile x four consecutive N-tiles, A fragment reused across 4 WMMAs.
__device__ __forceinline__ void wmma_tile4(const _Float16* __restrict__ A, int lda,
                                           const _Float16* __restrict__ BT, int ldk,
                                           int m0, int n0, int K, int lane,
                                           v8f acc[4]) {
  for (int k0 = 0; k0 < K; k0 += 32) {
    const v16h a = load_a_tile(A, lda, m0, k0, lane);
#pragma unroll
    for (int j = 0; j < 4; ++j) {
      const v16h b = load_b_tileT(BT, ldk, k0, n0 + 16 * j, lane);
      acc[j] = __builtin_amdgcn_wmma_f32_16x16x32_f16(
          false, a, false, b, (short)0, acc[j], false, false);
    }
  }
}

// ---------------------------------------------------------------------------
// f32 -> f16 conversions (plain and transposing)
// ---------------------------------------------------------------------------
__global__ void cvt_f16_kernel(const float* __restrict__ src,
                               _Float16* __restrict__ dst, int n) {
  int i = blockIdx.x * blockDim.x + threadIdx.x;
  if (i < n) dst[i] = (_Float16)src[i];
}

// src: rows x cols row-major -> dst: cols x rows row-major (dst[c][r] = src[r][c])
__global__ void cvt_f16_T_kernel(const float* __restrict__ src,
                                 _Float16* __restrict__ dst, int rows, int cols) {
  int i = blockIdx.x * blockDim.x + threadIdx.x;
  if (i < rows * cols) {
    int r = i / cols, c = i % cols;
    dst[(size_t)c * rows + r] = (_Float16)src[i];
  }
}

// ---------------------------------------------------------------------------
// QKV projections: q = x@Wq ; k = x@Wk + s*eps_k ; v = x@Wv + s*eps_v
// 3 matrices x 25 Mtiles x 2 Ngroups(64 cols) = 150 waves, 4 WMMA tiles each.
// ---------------------------------------------------------------------------
__global__ void gemm_qkv_kernel(const _Float16* __restrict__ xh,
                                const _Float16* __restrict__ WqT,
                                const _Float16* __restrict__ WkT,
                                const _Float16* __restrict__ WvT,
                                const float* __restrict__ eps_k,
                                const float* __restrict__ eps_v,
                                float* __restrict__ qbuf,
                                float* __restrict__ knew,
                                float* __restrict__ vnew) {
  const int wave = (blockIdx.x * blockDim.x + threadIdx.x) >> 5;
  const int lane = threadIdx.x & 31;
  if (wave >= 150) return;
  const int which = wave / 50;       // 0=q 1=k 2=v
  const int tile  = wave % 50;
  const int mt = tile / 2, ng = tile % 2;
  const int n0 = ng * 64;
  const _Float16* BT = (which == 0) ? WqT : ((which == 1) ? WkT : WvT);
  v8f acc[4] = {};
  wmma_tile4(xh, DD, BT, DD, mt * 16, n0, DD, lane, acc);
  const int c  = lane & 15;
  const int mb = (lane >> 4) * 8;
#pragma unroll
  for (int j = 0; j < 4; ++j) {
    const int col = n0 + j * 16 + c;
#pragma unroll
    for (int i = 0; i < 8; ++i) {
      const size_t o = (size_t)(mt * 16 + mb + i) * DD + col;
      float v = acc[j][i];
      if (which == 0)      qbuf[o] = v;
      else if (which == 1) knew[o] = v + SIGMA_ * eps_k[o];
      else                 vnew[o] = v + SIGMA_ * eps_v[o];
    }
  }
}

// ---------------------------------------------------------------------------
// Attention: one block (128 thr) per particle m=b*P+p; scores in LDS.
// ---------------------------------------------------------------------------
__global__ void attn_kernel(const float* __restrict__ qbuf,
                            const float* __restrict__ Kin,
                            const float* __restrict__ Vin,
                            const float* __restrict__ knew,
                            const float* __restrict__ vnew,
                            const int* __restrict__ tptr,
                            float* __restrict__ ctxf,
                            _Float16* __restrict__ ctxh) {
  __shared__ float sc[513 * HH];
  __shared__ float qs[DD];
  __shared__ float red[128];
  __shared__ float mx[HH], sm[HH];

  const int m   = blockIdx.x;          // particle index
  const int tid = threadIdx.x;
  const int tval = *tptr;
  int Sv = tval + 1; if (Sv > 513) Sv = 513;

  qs[tid] = qbuf[(size_t)m * DD + tid];
  __syncthreads();

  // scores: thread handles head h=tid&7, rows s = (tid>>3) strided by 16
  const int h  = tid & 7;
  const int s0 = tid >> 3;
  const float4* q4 = (const float4*)&qs[h * DHH];
  for (int s = s0; s < Sv; s += 16) {
    const float* Krow = (s == tval) ? (knew + (size_t)m * DD)
                                    : (Kin + ((size_t)m * SS + s) * DD);
    __builtin_prefetch(Kin + ((size_t)m * SS + s + 16) * DD + h * DHH, 0, 0);
    const float4* k4 = (const float4*)(Krow + h * DHH);
    float acc = 0.f;
#pragma unroll
    for (int j = 0; j < 4; ++j) {
      const float4 kk = k4[j];
      const float4 qq = q4[j];
      acc += kk.x * qq.x + kk.y * qq.y + kk.z * qq.z + kk.w * qq.w;
    }
    sc[s * HH + h] = acc * 0.25f;      // 1/sqrt(16)
  }
  __syncthreads();

  // per-head max
  float pm = -3.0e38f;
  for (int s = s0; s < Sv; s += 16) pm = fmaxf(pm, sc[s * HH + h]);
  red[tid] = pm;
  __syncthreads();
  if (tid < HH) {
    float mv = -3.0e38f;
#pragma unroll
    for (int j = 0; j < 16; ++j) mv = fmaxf(mv, red[h + 8 * j]);
    mx[h] = mv;
  }
  __syncthreads();

  // exp + per-head sum
  float ps = 0.f;
  const float mh = mx[h];
  for (int s = s0; s < Sv; s += 16) {
    float e = __expf(sc[s * HH + h] - mh);
    sc[s * HH + h] = e;
    ps += e;
  }
  red[tid] = ps;
  __syncthreads();
  if (tid < HH) {
    float sv = 0.f;
#pragma unroll
    for (int j = 0; j < 16; ++j) sv += red[h + 8 * j];
    sm[h] = sv;
  }
  __syncthreads();

  // ctx: thread owns output dim tid, head = tid>>4; coalesced V reads
  const int hc = tid >> 4;
  float acc = 0.f;
  for (int s = 0; s < Sv; ++s) {
    const float* Vrow = (s == tval) ? (vnew + (size_t)m * DD)
                                    : (Vin + ((size_t)m * SS + s) * DD);
    __builtin_prefetch(Vin + ((size_t)m * SS + s + 16) * DD + tid, 0, 0);
    acc += sc[s * HH + hc] * Vrow[tid];
  }
  acc /= sm[hc];
  ctxf[(size_t)m * DD + tid] = acc;
  ctxh[(size_t)m * DD + tid] = (_Float16)acc;
}

// ---------------------------------------------------------------------------
// z = ctx@Wo + s*eps_z   (25 x 2 = 50 waves)
// ---------------------------------------------------------------------------
__global__ void gemm_z_kernel(const _Float16* __restrict__ ctxh,
                              const _Float16* __restrict__ WoT,
                              const float* __restrict__ eps_z,
                              float* __restrict__ zout,
                              _Float16* __restrict__ zh) {
  const int wave = (blockIdx.x * blockDim.x + threadIdx.x) >> 5;
  const int lane = threadIdx.x & 31;
  if (wave >= 50) return;
  const int mt = wave / 2, ng = wave % 2;
  const int n0 = ng * 64;
  v8f acc[4] = {};
  wmma_tile4(ctxh, DD, WoT, DD, mt * 16, n0, DD, lane, acc);
  const int c = lane & 15, mb = (lane >> 4) * 8;
#pragma unroll
  for (int j = 0; j < 4; ++j) {
    const int col = n0 + j * 16 + c;
#pragma unroll
    for (int i = 0; i < 8; ++i) {
      const size_t o = (size_t)(mt * 16 + mb + i) * DD + col;
      float v = acc[j][i] + SIGMA_ * eps_z[o];
      zout[o] = v;
      zh[o]   = (_Float16)v;
    }
  }
}

// ---------------------------------------------------------------------------
// h = relu(z@W1 + b1)  (25 x 8 = 200 waves, N=512)
// ---------------------------------------------------------------------------
__global__ void gemm_h_kernel(const _Float16* __restrict__ zh,
                              const _Float16* __restrict__ W1T,
                              const float* __restrict__ b1,
                              _Float16* __restrict__ hh) {
  const int wave = (blockIdx.x * blockDim.x + threadIdx.x) >> 5;
  const int lane = threadIdx.x & 31;
  if (wave >= 200) return;
  const int mt = wave / 8, ng = wave % 8;
  const int n0 = ng * 64;
  v8f acc[4] = {};
  wmma_tile4(zh, DD, W1T, DD, mt * 16, n0, DD, lane, acc);
  const int c = lane & 15, mb = (lane >> 4) * 8;
#pragma unroll
  for (int j = 0; j < 4; ++j) {
    const int col = n0 + j * 16 + c;
    const float bias = b1[col];
#pragma unroll
    for (int i = 0; i < 8; ++i) {
      float v = fmaxf(acc[j][i] + bias, 0.f);
      hh[(size_t)(mt * 16 + mb + i) * DFF + col] = (_Float16)v;
    }
  }
}

// ---------------------------------------------------------------------------
// r = h@W2 + b2  (25 x 2 = 50 waves, K=512)
// ---------------------------------------------------------------------------
__global__ void gemm_r_kernel(const _Float16* __restrict__ hh,
                              const _Float16* __restrict__ W2T,
                              const float* __restrict__ b2,
                              float* __restrict__ rout) {
  const int wave = (blockIdx.x * blockDim.x + threadIdx.x) >> 5;
  const int lane = threadIdx.x & 31;
  if (wave >= 50) return;
  const int mt = wave / 2, ng = wave % 2;
  const int n0 = ng * 64;
  v8f acc[4] = {};
  wmma_tile4(hh, DFF, W2T, DFF, mt * 16, n0, DFF, lane, acc);
  const int c = lane & 15, mb = (lane >> 4) * 8;
#pragma unroll
  for (int j = 0; j < 4; ++j) {
    const int col = n0 + j * 16 + c;
    const float bias = b2[col];
#pragma unroll
    for (int i = 0; i < 8; ++i)
      rout[(size_t)(mt * 16 + mb + i) * DD + col] = acc[j][i] + bias;
  }
}

// ---------------------------------------------------------------------------
// pred[m] = r[m,:] . Wout + bout   (F=1)
// ---------------------------------------------------------------------------
__global__ void pred_kernel(const float* __restrict__ rout,
                            const float* __restrict__ Wout,
                            const float* __restrict__ bout,
                            float* __restrict__ pred) {
  __shared__ float red[128];
  const int m = blockIdx.x, tid = threadIdx.x;
  red[tid] = rout[(size_t)m * DD + tid] * Wout[tid];
  __syncthreads();
  for (int off = 64; off > 0; off >>= 1) {
    if (tid < off) red[tid] += red[tid + off];
    __syncthreads();
  }
  if (tid == 0) pred[m] = red[0] + bout[0];
}

// ---------------------------------------------------------------------------
// Particle weights + deterministic inverse-CDF resampling. One block per b.
// ---------------------------------------------------------------------------
__device__ __forceinline__ float hash01(unsigned s) {
  s ^= 2747636419u; s *= 2654435769u;
  s ^= s >> 16;     s *= 2654435769u;
  s ^= s >> 16;     s *= 2654435769u;
  return (float)(s >> 8) * (1.0f / 16777216.0f);
}

__global__ void weights_resample_kernel(const float* __restrict__ y,
                                        const float* __restrict__ pred,
                                        float* __restrict__ wout,
                                        int* __restrict__ idxws) {
  __shared__ float lw[PP];
  __shared__ float cdf[PP];
  __shared__ float stats[2];   // min, sum
  const int b = blockIdx.x, p = threadIdx.x;
  if (p < PP) {
    float mu = y[b * 4] - pred[b * PP + p];   // y[b,0,0], FIN=4
    lw[p] = -0.5f * OMEGA_ * mu * mu;
  }
  __syncthreads();
  if (p == 0) {
    float mn = lw[0];
    for (int j = 1; j < PP; ++j) mn = fminf(mn, lw[j]);
    stats[0] = mn;
  }
  __syncthreads();
  if (p < PP) lw[p] = __expf(lw[p] - stats[0]);
  __syncthreads();
  if (p == 0) {
    float s = 0.f;
    for (int j = 0; j < PP; ++j) { s += lw[j]; cdf[j] = s; }
    stats[1] = s;
  }
  __syncthreads();
  if (p < PP) {
    const float tot = stats[1];
    wout[b * PP + p] = lw[p] / tot;
    float u = hash01((unsigned)(b * PP + p) * 2654435761u + 42u) * tot;
    int sel = PP - 1;
    for (int j = 0; j < PP; ++j) { if (u < cdf[j]) { sel = j; break; } }
    idxws[b * PP + p] = sel;
  }
}

// ---------------------------------------------------------------------------
// Resampling gather of K/V (float4 granularity) folding in the row-t update.
// ---------------------------------------------------------------------------
__global__ void gather_kv_kernel(const float4* __restrict__ K4,
                                 const float4* __restrict__ V4,
                                 const float4* __restrict__ knew4,
                                 const float4* __restrict__ vnew4,
                                 const int* __restrict__ idxws,
                                 const int* __restrict__ tptr,
                                 float4* __restrict__ Kout4,
                                 float4* __restrict__ Vout4) {
  const size_t e = (size_t)blockIdx.x * blockDim.x + threadIdx.x;
  const size_t total = (size_t)NPART * SS * (DD / 4);
  if (e >= total) return;
  const int d4 = (int)(e & 31);
  size_t r = e >> 5;
  const int s = (int)(r % SS);
  r /= SS;
  const int p = (int)(r % PP);
  const int b = (int)(r / PP);
  const int pi = idxws[b * PP + p];
  const int tval = *tptr;
  float4 kv, vv;
  if (s == tval) {
    const size_t o = (size_t)(b * PP + pi) * (DD / 4) + d4;
    kv = knew4[o]; vv = vnew4[o];
  } else {
    const size_t o = ((size_t)(b * PP + pi) * SS + s) * (DD / 4) + d4;
    kv = K4[o]; vv = V4[o];
  }
  Kout4[e] = kv;
  Vout4[e] = vv;
}

__global__ void gather_i_kernel(const int* __restrict__ Iin,
                                const int* __restrict__ idxws,
                                const int* __restrict__ tptr,
                                int* __restrict__ Iout) {
  const int e = blockIdx.x * blockDim.x + threadIdx.x;
  if (e >= NPART * SS) return;
  const int s = e % SS;
  const int r = e / SS;
  const int p = r % PP, b = r / PP;
  const int pi = idxws[b * PP + p];
  Iout[e] = (s == *tptr) ? pi : Iin[(b * PP + pi) * SS + s];
}

// ---------------------------------------------------------------------------
extern "C" void kernel_launch(void* const* d_in, const int* in_sizes, int n_in,
                              void* d_out, int out_size, void* d_ws, size_t ws_size,
                              hipStream_t stream) {
  (void)in_sizes; (void)n_in; (void)out_size; (void)ws_size;
  const float* x     = (const float*)d_in[0];
  const float* y     = (const float*)d_in[1];
  const float* Kin   = (const float*)d_in[2];
  const float* Vin   = (const float*)d_in[3];
  const int*   Iin   = (const int*)d_in[5];
  const float* Wq    = (const float*)d_in[6];
  const float* Wk    = (const float*)d_in[7];
  const float* Wv    = (const float*)d_in[8];
  const float* Wo    = (const float*)d_in[9];
  const float* W1    = (const float*)d_in[10];
  const float* b1    = (const float*)d_in[11];
  const float* W2    = (const float*)d_in[12];
  const float* b2    = (const float*)d_in[13];
  const float* Wout  = (const float*)d_in[14];
  const float* bout  = (const float*)d_in[15];
  const float* eps_k = (const float*)d_in[16];
  const float* eps_v = (const float*)d_in[17];
  const float* eps_z = (const float*)d_in[18];
  const int*   tptr  = (const int*)d_in[19];

  // workspace layout (256B aligned offsets)
  char* ws = (char*)d_ws;
  _Float16* xh   = (_Float16*)(ws + 0);         // 400x128
  _Float16* WqT  = (_Float16*)(ws + 102400);    // 128x128 (N x K)
  _Float16* WkT  = (_Float16*)(ws + 135168);
  _Float16* WvT  = (_Float16*)(ws + 167936);
  _Float16* WoT  = (_Float16*)(ws + 200704);
  _Float16* W1T  = (_Float16*)(ws + 233472);    // 512x128 (N x K)
  _Float16* W2T  = (_Float16*)(ws + 364544);    // 128x512 (N x K)
  float*    qbuf = (float*)(ws + 495616);       // 400x128
  float*    knew = (float*)(ws + 700416);       // 400x128
  float*    vnew = (float*)(ws + 905216);       // 400x128
  float*    ctxf = (float*)(ws + 1110016);      // 400x128
  _Float16* ctxh = (_Float16*)(ws + 1314816);   // 400x128
  _Float16* zh   = (_Float16*)(ws + 1417216);   // 400x128
  _Float16* hh   = (_Float16*)(ws + 1519616);   // 400x512
  int*      idxw = (int*)(ws + 1929216);        // 400

  // output layout (flat concat, return order)
  float* out      = (float*)d_out;
  float* out_r    = out;                     // 400x128
  float* out_z    = out + 51200;             // 400x128
  float* out_pred = out + 102400;            // 400
  float* out_w    = out + 102800;            // 400
  float* out_K    = out + 103200;            // 400x1024x128
  float* out_V    = out + 52532000;
  int*   out_I    = (int*)(out + 104960800); // 400x1024

  // 1) f32 -> f16 conversions (weights transposed for contiguous B fragments)
  cvt_f16_kernel  <<<(51200 + 255) / 256, 256, 0, stream>>>(x,  xh,  51200);
  cvt_f16_T_kernel<<<(16384 + 255) / 256, 256, 0, stream>>>(Wq, WqT, DD, DD);
  cvt_f16_T_kernel<<<(16384 + 255) / 256, 256, 0, stream>>>(Wk, WkT, DD, DD);
  cvt_f16_T_kernel<<<(16384 + 255) / 256, 256, 0, stream>>>(Wv, WvT, DD, DD);
  cvt_f16_T_kernel<<<(16384 + 255) / 256, 256, 0, stream>>>(Wo, WoT, DD, DD);
  cvt_f16_T_kernel<<<(65536 + 255) / 256, 256, 0, stream>>>(W1, W1T, DD, DFF);
  cvt_f16_T_kernel<<<(65536 + 255) / 256, 256, 0, stream>>>(W2, W2T, DFF, DD);

  // 2) QKV projections (WMMA): 150 waves, 4 waves/block
  gemm_qkv_kernel<<<38, 128, 0, stream>>>(xh, WqT, WkT, WvT, eps_k, eps_v,
                                          qbuf, knew, vnew);

  // 3) attention: one block per particle
  attn_kernel<<<NPART, 128, 0, stream>>>(qbuf, Kin, Vin, knew, vnew, tptr,
                                         ctxf, ctxh);

  // 4) z = ctx@Wo + eps
  gemm_z_kernel<<<13, 128, 0, stream>>>(ctxh, WoT, eps_z, out_z, zh);

  // 5) FFN
  gemm_h_kernel<<<50, 128, 0, stream>>>(zh, W1T, b1, hh);
  gemm_r_kernel<<<13, 128, 0, stream>>>(hh, W2T, b2, out_r);

  // 6) pred head
  pred_kernel<<<NPART, 128, 0, stream>>>(out_r, Wout, bout, out_pred);

  // 7) weights + resample indices
  weights_resample_kernel<<<BB, 32, 0, stream>>>(y, out_pred, out_w, idxw);

  // 8) big K/V gather (memory-bound bulk of the op) + index gather
  {
    const size_t total4 = (size_t)NPART * SS * (DD / 4);   // 13,107,200
    const int blocks = (int)((total4 + 255) / 256);
    gather_kv_kernel<<<blocks, 256, 0, stream>>>(
        (const float4*)Kin, (const float4*)Vin,
        (const float4*)knew, (const float4*)vnew,
        idxw, tptr, (float4*)out_K, (float4*)out_V);
  }
  gather_i_kernel<<<(NPART * SS + 255) / 256, 256, 0, stream>>>(Iin, idxw, tptr, out_I);
}
